// SimpleMultiheadSDPAttention_26465588478246
// MI455X (gfx1250) — matrix operand order repaired and checked by reading, not compiled, not run on hardware
//
#include <hip/hip_runtime.h>
#include <hip/hip_bf16.h>
#include <stdint.h>

// ---------------------------------------------------------------------------
// CDNA5 (gfx1250) WMMA multi-head causal attention.
//  - All matmuls via v_wmma_f32_16x16x32_f16 (f16 inputs, f32 accumulate)
//  - wave32, 256-thread blocks = 8 waves
//  - GEMM weight staging via global_load_async_to_lds_b128 (ASYNCcnt-tracked)
// ---------------------------------------------------------------------------

typedef __attribute__((ext_vector_type(16))) _Float16 v16h;
typedef __attribute__((ext_vector_type(8)))  float    v8f;

#if defined(__has_builtin)
#if __has_builtin(__builtin_amdgcn_global_load_async_to_lds_b128) && \
    __has_builtin(__builtin_amdgcn_s_wait_asynccnt)
#define HAVE_ASYNC_LDS 1
#endif
#endif
#ifndef HAVE_ASYNC_LDS
#define HAVE_ASYNC_LDS 0
#endif

#if HAVE_ASYNC_LDS
// Signature (from clang diagnostic): param0 = v4i32 in AS1 (global),
// param1 = LDS pointer (AS3), then two immediate ints (offset, cpol).
typedef int v4i_async __attribute__((vector_size(4 * sizeof(int))));
#define ASYNC_GPTR(p) ((__attribute__((address_space(1))) v4i_async*)(p))
#define ASYNC_LPTR(p) ((__attribute__((address_space(3))) v4i_async*)(p))
#endif

union Frag32B { uint4 u[2]; v16h v; };

// A-operand fragment (16x32 f16). Per ISA layout: lane m = lane&15 holds
// K = {lh*8 .. lh*8+7} and {16+lh*8 .. 16+lh*8+7}, lh = lane>>4.
__device__ __forceinline__ v16h load_a_frag(const _Float16* rowBase, int lh) {
  Frag32B f;
  f.u[0] = *(const uint4*)(rowBase + lh * 8);
  f.u[1] = *(const uint4*)(rowBase + 16 + lh * 8);
  return f.v;
}

// B-operand fragment (32x16 f16) for NT gemm: B[k][n] = Wrow_n[k].
// Lane n = lane&15 holds K = {lh*8 .. lh*8+7} and {16+lh*8 .. 16+lh*8+7}.
__device__ __forceinline__ v16h load_b_frag(const _Float16* rowBase, int lh) {
  Frag32B f;
  f.u[0] = *(const uint4*)(rowBase + lh * 8);
  f.u[1] = *(const uint4*)(rowBase + 16 + lh * 8);
  return f.v;
}

// ---------------------------------------------------------------------------
// fp32 -> f16 weight conversion
// ---------------------------------------------------------------------------
__global__ void cvt_f32_f16(const float* __restrict__ src,
                            _Float16* __restrict__ dst, int n) {
  int i = blockIdx.x * 256 + threadIdx.x;
  if (i < n) dst[i] = (_Float16)src[i];
}

// ---------------------------------------------------------------------------
// NT GEMM: Out[m,n] = sum_k A[m,k] * W[n,k] + bias[n]
//   A: fp32 (A_IS_F32) or f16, row-major [M,K];  W: f16, row-major [N,K]
// OUT_MODE 0: f16 out, [B,H,L,64] layout   (Q, K)
// OUT_MODE 1: f16 out, [B,H,64,L] layout   (V transposed)
// OUT_MODE 2: fp32 out, row-major [M,N]    (final projection)
// Block tile 128x128, K step 32, 8 waves each computing 32x64.
// ---------------------------------------------------------------------------
template <int OUT_MODE, bool A_IS_F32>
__global__ __launch_bounds__(256) void gemm_nt_wmma(
    const void* __restrict__ Av, const _Float16* __restrict__ W,
    const float* __restrict__ bias, void* __restrict__ Out,
    int M, int N, int K) {
  constexpr int LDT = 40;  // padded LDS row stride (f16 elems), 80B (16B-aligned)
  __shared__ __align__(16) _Float16 As[128 * LDT];
  __shared__ __align__(16) _Float16 Bs[128 * LDT];

  const int tid  = threadIdx.x;
  const int lane = tid & 31;
  const int wave = tid >> 5;
  const int ln   = lane & 15;
  const int lh   = lane >> 4;
  const int wm   = wave & 3;   // wave row: 4 x 32
  const int wn   = wave >> 2;  // wave col: 2 x 64
  const int m0   = blockIdx.y * 128;
  const int n0   = blockIdx.x * 128;

  const int sr = tid >> 1;  // staging row 0..127
  const int sh = tid & 1;   // staging half (16 elems)

  v8f zero = {0.f, 0.f, 0.f, 0.f, 0.f, 0.f, 0.f, 0.f};
  v8f acc[2][4];
#pragma unroll
  for (int mi = 0; mi < 2; mi++)
#pragma unroll
    for (int ni = 0; ni < 4; ni++) acc[mi][ni] = zero;

  for (int k0 = 0; k0 < K; k0 += 32) {
    __syncthreads();
    // ---- stage W tile (128x32, already f16): async copy straight to LDS ----
    {
      const _Float16* gp = W + (size_t)(n0 + sr) * K + k0 + sh * 16;
      _Float16*       lp = &Bs[sr * LDT + sh * 16];
#if HAVE_ASYNC_LDS
      __builtin_amdgcn_global_load_async_to_lds_b128(
          ASYNC_GPTR(gp), ASYNC_LPTR(lp), 0, 0);
      __builtin_amdgcn_global_load_async_to_lds_b128(
          ASYNC_GPTR(gp + 8), ASYNC_LPTR(lp + 8), 0, 0);
#else
      const uint4* b4 = (const uint4*)gp;
      uint4 x0 = b4[0], x1 = b4[1];
      *(uint4*)lp       = x0;
      *(uint4*)(lp + 8) = x1;
#endif
    }
    // ---- stage A tile (128x32) into LDS as f16 (conversion path) ----
    {
      Frag32B fr;
      _Float16* h = (_Float16*)&fr;
      if (A_IS_F32) {
        const float* A  = (const float*)Av;
        const float* ap = A + (size_t)(m0 + sr) * K + k0 + sh * 16;
#pragma unroll
        for (int q = 0; q < 4; q++) {
          float4 f = ((const float4*)ap)[q];
          h[q * 4 + 0] = (_Float16)f.x;
          h[q * 4 + 1] = (_Float16)f.y;
          h[q * 4 + 2] = (_Float16)f.z;
          h[q * 4 + 3] = (_Float16)f.w;
        }
        if (k0 + 32 < K) __builtin_prefetch(ap + 32, 0, 0);
      } else {
        const _Float16* A  = (const _Float16*)Av;
        const uint4*    a4 = (const uint4*)(A + (size_t)(m0 + sr) * K + k0 + sh * 16);
        fr.u[0] = a4[0];
        fr.u[1] = a4[1];
        if (k0 + 32 < K) __builtin_prefetch((const char*)a4 + 64, 0, 0);
      }
      *(uint4*)&As[sr * LDT + sh * 16]     = fr.u[0];
      *(uint4*)&As[sr * LDT + sh * 16 + 8] = fr.u[1];
    }
#if HAVE_ASYNC_LDS
    __builtin_amdgcn_s_wait_asynccnt(0);
#endif
    __syncthreads();

    v16h af[2], bf[4];
#pragma unroll
    for (int mi = 0; mi < 2; mi++)
      af[mi] = load_a_frag(&As[(wm * 32 + mi * 16 + ln) * LDT], lh);
#pragma unroll
    for (int ni = 0; ni < 4; ni++)
      bf[ni] = load_b_frag(&Bs[(wn * 64 + ni * 16 + ln) * LDT], lh);

#pragma unroll
    for (int mi = 0; mi < 2; mi++)
#pragma unroll
      for (int ni = 0; ni < 4; ni++)
        acc[mi][ni] = __builtin_amdgcn_wmma_f32_16x16x32_f16(
            false, af[mi], false, bf[ni], (short)0, acc[mi][ni], false, false);
  }

  // ---- epilogue: bias + scatter per mode ----
#pragma unroll
  for (int mi = 0; mi < 2; mi++)
#pragma unroll
    for (int ni = 0; ni < 4; ni++) {
      const int   n  = n0 + wn * 64 + ni * 16 + ln;
      const float bv = bias[n];
#pragma unroll
      for (int r = 0; r < 8; r++) {
        const int   m   = m0 + wm * 32 + mi * 16 + r + lh * 8;
        const float val = acc[mi][ni][r] + bv;
        if (OUT_MODE == 2) {
          ((float*)Out)[(size_t)m * N + n] = val;
        } else {
          const int bb = m >> 11, l = m & 2047;  // L = 2048
          const int hh = n >> 6,  d = n & 63;    // Dh = 64
          _Float16* O = (_Float16*)Out;
          if (OUT_MODE == 0)
            O[(((size_t)(bb * 16 + hh)) * 2048 + l) * 64 + d] = (_Float16)val;
          else
            O[(((size_t)(bb * 16 + hh)) * 64 + d) * 2048 + l] = (_Float16)val;
        }
      }
    }
}

// ---------------------------------------------------------------------------
// Causal flash attention, one (b,h) per blockIdx.y, 128 q-rows per block,
// 16 q-rows per wave. kv tiles of 32. Q/K: [B,H,L,64] f16, V: [B,H,64,L] f16.
// Softmax runs in the exp2 domain (scores pre-scaled by 0.125*log2(e)).
// Output: f16 [B, L, 1024] (concat heads) for the final projection.
// ---------------------------------------------------------------------------
__global__ __launch_bounds__(256) void attn_causal_wmma(
    const _Float16* __restrict__ Qw, const _Float16* __restrict__ Kw,
    const _Float16* __restrict__ Vt, _Float16* __restrict__ Aout) {
  __shared__ __align__(16) _Float16 Pl[8][16 * 32];  // per-wave P staging

  const int tid  = threadIdx.x;
  const int lane = tid & 31;
  const int wave = tid >> 5;
  const int ln   = lane & 15;
  const int lh   = lane >> 4;
  const int bh   = blockIdx.y;
  const int b    = bh >> 4;
  const int h    = bh & 15;
  const int qBase = blockIdx.x * 128 + wave * 16;

  const float C = 0.125f * 1.44269504088896340736f;  // 1/sqrt(Dh) * log2(e)

  const _Float16* Qb = Qw + (size_t)bh * 2048 * 64;
  const _Float16* Kb = Kw + (size_t)bh * 2048 * 64;
  const _Float16* Vb = Vt + (size_t)bh * 64 * 2048;

  v16h qf[2];
#pragma unroll
  for (int c = 0; c < 2; c++)
    qf[c] = load_a_frag(Qb + (size_t)(qBase + ln) * 64 + c * 32, lh);

  v8f zero = {0.f, 0.f, 0.f, 0.f, 0.f, 0.f, 0.f, 0.f};
  v8f o[4] = {zero, zero, zero, zero};
  float m8[8], l8[8];
#pragma unroll
  for (int r = 0; r < 8; r++) { m8[r] = -3.0e38f; l8[r] = 0.f; }

  // one kv tile (32 columns); `masked` constant-folds at each call site
  auto kv_tile = [&](int kv0, bool masked) __attribute__((always_inline)) {
    // ---- S = (Q K^T) * C (log2-domain), optional causal mask ----
    v8f st[2];
#pragma unroll
    for (int j = 0; j < 2; j++) {
      v16h kb0 = load_b_frag(Kb + (size_t)(kv0 + j * 16 + ln) * 64, lh);
      v16h kb1 = load_b_frag(Kb + (size_t)(kv0 + j * 16 + ln) * 64 + 32, lh);
      v8f s = __builtin_amdgcn_wmma_f32_16x16x32_f16(
          false, qf[0], false, kb0, (short)0, zero, false, false);
      s = __builtin_amdgcn_wmma_f32_16x16x32_f16(
          false, qf[1], false, kb1, (short)0, s, false, false);
      const int col = kv0 + j * 16 + ln;
#pragma unroll
      for (int r = 0; r < 8; r++) {
        const float x = s[r] * C;
        if (masked) {
          const int row = qBase + r + lh * 8;
          st[j][r] = (col > row) ? -1.0e30f : x;
        } else {
          st[j][r] = x;
        }
      }
    }

    // ---- row max (4 xor-shuffles within 16-lane halves) ----
    float rmax[8];
#pragma unroll
    for (int r = 0; r < 8; r++) {
      float mx = fmaxf(st[0][r], st[1][r]);
      mx = fmaxf(mx, __shfl_xor(mx, 1, 32));
      mx = fmaxf(mx, __shfl_xor(mx, 2, 32));
      mx = fmaxf(mx, __shfl_xor(mx, 4, 32));
      mx = fmaxf(mx, __shfl_xor(mx, 8, 32));
      rmax[r] = mx;
    }

    // ---- online softmax (exp2 domain): rescale, exponentiate, row sum ----
#pragma unroll
    for (int r = 0; r < 8; r++) {
      const float mn = fmaxf(m8[r], rmax[r]);
      const float sc = exp2f(m8[r] - mn);
      m8[r] = mn;
      l8[r] *= sc;
#pragma unroll
      for (int sub = 0; sub < 4; sub++) o[sub][r] *= sc;
      const float p0 = exp2f(st[0][r] - mn);
      const float p1 = exp2f(st[1][r] - mn);
      st[0][r] = p0;
      st[1][r] = p1;
      float rs = p0 + p1;
      rs += __shfl_xor(rs, 1, 32);
      rs += __shfl_xor(rs, 2, 32);
      rs += __shfl_xor(rs, 4, 32);
      rs += __shfl_xor(rs, 8, 32);
      l8[r] += rs;
    }

    // ---- C-layout -> A-layout via per-wave LDS patch ----
    _Float16* P = &Pl[wave][0];
#pragma unroll
    for (int j = 0; j < 2; j++)
#pragma unroll
      for (int r = 0; r < 8; r++)
        P[(r + lh * 8) * 32 + j * 16 + ln] = (_Float16)st[j][r];
    // same-wave LDS ops are in-order; regions are wave-private (no barrier)
    v16h pa = load_a_frag(P + ln * 32, lh);

    // ---- O += P * V (V stored transposed: contiguous B-fragments) ----
#pragma unroll
    for (int sub = 0; sub < 4; sub++) {
      v16h vb = load_b_frag(Vb + (size_t)(sub * 16 + ln) * 2048 + kv0, lh);
      o[sub] = __builtin_amdgcn_wmma_f32_16x16x32_f16(
          false, pa, false, vb, (short)0, o[sub], false, false);
    }
  };

  // tiles with kv0 + 31 <= qBase need no causal mask
  const int ntFull = qBase >> 5;
  const int nt     = (qBase + 16 + 31) >> 5;
  for (int kt = 0; kt < ntFull; kt++) kv_tile(kt * 32, false);
  for (int kt = ntFull; kt < nt; kt++) kv_tile(kt * 32, true);

  // ---- epilogue: O / l, scatter heads back to [B, L, 1024] f16 ----
#pragma unroll
  for (int sub = 0; sub < 4; sub++)
#pragma unroll
    for (int r = 0; r < 8; r++) {
      const int row = qBase + r + lh * 8;
      const int e   = h * 64 + sub * 16 + ln;
      Aout[((size_t)(b * 2048 + row)) * 1024 + e] = (_Float16)(o[sub][r] / l8[r]);
    }
}

// ---------------------------------------------------------------------------
// Host-side orchestration
// ---------------------------------------------------------------------------
extern "C" void kernel_launch(void* const* d_in, const int* in_sizes, int n_in,
                              void* d_out, int out_size, void* d_ws, size_t ws_size,
                              hipStream_t stream) {
  (void)in_sizes; (void)n_in; (void)out_size; (void)ws_size;
  const float* q_in = (const float*)d_in[0];
  const float* k_in = (const float*)d_in[1];
  const float* v_in = (const float*)d_in[2];
  const float* Wq   = (const float*)d_in[3];
  const float* bq   = (const float*)d_in[4];
  const float* Wk   = (const float*)d_in[5];
  const float* bk   = (const float*)d_in[6];
  const float* Wv   = (const float*)d_in[7];
  const float* bv   = (const float*)d_in[8];
  const float* Wo   = (const float*)d_in[9];
  const float* bo   = (const float*)d_in[10];

  char* ws = (char*)d_ws;
  const size_t MB = 1024 * 1024;
  _Float16* WqH = (_Float16*)(ws + 0 * MB);   // 2 MiB each
  _Float16* WkH = (_Float16*)(ws + 2 * MB);
  _Float16* WvH = (_Float16*)(ws + 4 * MB);
  _Float16* WoH = (_Float16*)(ws + 6 * MB);
  _Float16* Qw  = (_Float16*)(ws + 8 * MB);   // 16 MiB each
  _Float16* Kw  = (_Float16*)(ws + 24 * MB);
  _Float16* Vt  = (_Float16*)(ws + 40 * MB);
  _Float16* At  = (_Float16*)(ws + 56 * MB);  // end: 72 MiB

  const int WE = 1024 * 1024;
  cvt_f32_f16<<<WE / 256, 256, 0, stream>>>(Wq, WqH, WE);
  cvt_f32_f16<<<WE / 256, 256, 0, stream>>>(Wk, WkH, WE);
  cvt_f32_f16<<<WE / 256, 256, 0, stream>>>(Wv, WvH, WE);
  cvt_f32_f16<<<WE / 256, 256, 0, stream>>>(Wo, WoH, WE);

  const dim3 gg(1024 / 128, 8192 / 128);  // (N tiles, M tiles) = (8, 64)
  const dim3 blk(256);
  gemm_nt_wmma<0, true><<<gg, blk, 0, stream>>>(q_in, WqH, bq, Qw, 8192, 1024, 1024);
  gemm_nt_wmma<0, true><<<gg, blk, 0, stream>>>(k_in, WkH, bk, Kw, 8192, 1024, 1024);
  gemm_nt_wmma<1, true><<<gg, blk, 0, stream>>>(v_in, WvH, bv, Vt, 8192, 1024, 1024);

  attn_causal_wmma<<<dim3(2048 / 128, 4 * 16), blk, 0, stream>>>(Qw, Kw, Vt, At);

  gemm_nt_wmma<2, false><<<gg, blk, 0, stream>>>(At, WoH, bo, d_out, 8192, 1024, 1024);
}
